// RowColumnAttention_81801947119688
// MI455X (gfx1250) — compile-verified, matrix operand order checked by reading, not run on hardware
//
#include <hip/hip_runtime.h>
#include <hip/hip_bf16.h>
#include <math.h>
#include <stdint.h>

typedef __attribute__((ext_vector_type(16))) _Float16 v16h;
typedef __attribute__((ext_vector_type(8)))  _Float16 v8h;
typedef __attribute__((ext_vector_type(8)))  float    v8f;

static constexpr int kH    = 1024;
static constexpr int kNH   = 16;
static constexpr int kNKV  = 8;
static constexpr int kDH   = 128;
static constexpr int kFF   = 3072;
static constexpr int kB    = 128;
static constexpr int kT    = 16;
static constexpr int kL    = 128;
static constexpr int kSrow = kT + kL;   // 144
static constexpr int SMAXP = 160;       // padded S (mult of 32), Vt row stride
static constexpr float kEps   = 1e-6f;
static constexpr float kNeg   = -3.4028234663852886e38f;
static constexpr float kScale = 0.08838834764831845f;   // DH^-0.5
static constexpr float kLnTheta = 13.815510557964274f;  // ln(1e6)

// ---------------------------------------------------------------------------
// WMMA helpers (gfx1250, wave32). 16x16x32 f16 -> f32 accumulate.
// A-fragment (16xK tile, row-major source, stride ld in halfs):
//   lanes 0-15 : row = lane,    elems 0-7 = K 0-7,  elems 8-15 = K 16-23
//   lanes 16-31: row = lane-16, elems 0-7 = K 8-15, elems 8-15 = K 24-31
// B-fragment from an N-major (transposed) source has the identical pattern
// with "row" = output column n. Both load as two contiguous 16B chunks.
// ---------------------------------------------------------------------------
__device__ __forceinline__ v16h frag_ld(const _Float16* base, int ld) {
  int lane = threadIdx.x & 31;
  int r  = lane & 15;
  int ko = (lane >> 4) << 3;              // 0 or 8
  const _Float16* p = base + (size_t)r * ld + ko;
  v8h lo = *(const v8h*)(p);
  v8h hi = *(const v8h*)(p + 16);
  return __builtin_shufflevector(lo, hi, 0,1,2,3,4,5,6,7,8,9,10,11,12,13,14,15);
}

__device__ __forceinline__ v8f wmma32(v16h a, v16h b, v8f c) {
  return __builtin_amdgcn_wmma_f32_16x16x32_f16(false, a, false, b, (short)0, c, false, false);
}

// ---------------------------------------------------------------------------
// Tiled WMMA GEMM: C[M,N] = A[M,K] * B[K,N], A/B f16 row-major, C f32 or f16.
// Block 256 thr = 8 waves (2x4). Block tile 64x128, wave tile 32x32, K-step 32.
// A tile staged with CDNA5 async global->LDS DMA (ASYNCcnt path, bypasses
// VGPRs); B tile transposed into LDS manually so B-fragments load contiguous.
// M%64==0, N%128==0, K%32==0 guaranteed by problem dims.
// ---------------------------------------------------------------------------
static constexpr int GTM = 64, GTN = 128, GTK = 32, GLD = GTK + 8; // 40 halfs

template<int OUTF16>
__global__ __launch_bounds__(256) void k_gemm(const _Float16* __restrict__ A,
                                              const _Float16* __restrict__ Bw,
                                              float* __restrict__ Cf,
                                              _Float16* __restrict__ Ch,
                                              int M, int N, int K) {
  __shared__ __align__(16) _Float16 sA[GTM * GLD];
  __shared__ __align__(16) _Float16 sB[GTN * GLD];
  const int tid  = threadIdx.x;
  const int wave = tid >> 5, lane = tid & 31;
  const int wm = wave >> 2, wn = wave & 3;
  const int m0 = blockIdx.y * GTM, n0 = blockIdx.x * GTN;

  v8f acc[2][2] = {};
  for (int k0 = 0; k0 < K; k0 += GTK) {
    { // stage A: 64x32 halfs = 4KB; one async b128 per lane, straight to LDS
      int idx = tid * 8;
      int r = idx >> 5, c = idx & 31;
      const _Float16* gp = A + (size_t)(m0 + r) * K + k0 + c;
      unsigned ldsOff = (unsigned)(uintptr_t)(&sA[r * GLD + c]);
      asm volatile("global_load_async_to_lds_b128 %0, %1, off"
                   :: "v"(ldsOff), "v"(gp) : "memory");
    }
    { // stage B: 32x128 halfs, transposed into sB[n][k] (overlaps async A)
#pragma unroll
      for (int part = 0; part < 2; ++part) {
        int idx = tid * 8 + part * 2048;
        int r = idx >> 7, c = idx & 127;             // r = k row, c = n col
        v8h d = *(const v8h*)(Bw + (size_t)(k0 + r) * N + n0 + c);
#pragma unroll
        for (int j = 0; j < 8; ++j) sB[(c + j) * GLD + r] = d[j];
      }
    }
    if (k0 + GTK < K) {  // global_prefetch_b8 of the next K tile
      __builtin_prefetch(A + (size_t)(m0 + (tid >> 2)) * K + k0 + GTK, 0, 1);
      __builtin_prefetch(Bw + (size_t)(k0 + GTK + (tid >> 3)) * N + n0, 0, 1);
    }
    asm volatile("s_wait_asynccnt 0x0" ::: "memory");   // own async A done
    __syncthreads();                                    // all waves' tiles done
    v16h a0 = frag_ld(sA + (wm * 32)      * GLD, GLD);
    v16h a1 = frag_ld(sA + (wm * 32 + 16) * GLD, GLD);
    v16h b0 = frag_ld(sB + (wn * 32)      * GLD, GLD);
    v16h b1 = frag_ld(sB + (wn * 32 + 16) * GLD, GLD);
    acc[0][0] = wmma32(a0, b0, acc[0][0]);
    acc[0][1] = wmma32(a0, b1, acc[0][1]);
    acc[1][0] = wmma32(a1, b0, acc[1][0]);
    acc[1][1] = wmma32(a1, b1, acc[1][1]);
    __syncthreads();
  }
  // epilogue: VGPR e -> row (tile + e + 8*(lane>=16)), col (tile + lane&15)
#pragma unroll
  for (int i = 0; i < 2; ++i)
#pragma unroll
    for (int j = 0; j < 2; ++j) {
      int row0 = m0 + wm * 32 + i * 16 + ((lane >> 4) << 3);
      int col  = n0 + wn * 32 + j * 16 + (lane & 15);
#pragma unroll
      for (int e = 0; e < 8; ++e) {
        size_t off = (size_t)(row0 + e) * N + col;
        if (OUTF16) Ch[off] = (_Float16)acc[i][j][e];
        else        Cf[off] = acc[i][j][e];
      }
    }
}

// ---------------------------------------------------------------------------
// Attention: one block (128 thr = 4 waves) per (head, batch).
// Q:[b][h][s][d] f16, K:[b][kv][s][d] f16, Vt:[b][kv][d][SMAXP] f16.
// Each wave owns 16 query rows: scores (WMMA) -> LDS f32 -> in-wave softmax
// -> P f16 in LDS -> P·V (WMMA) -> O[b][s][h*DH+d] f16.
// Mask: blocked iff (i>=16 && j>=16 && j>i)  (never triggers for col layer).
// ---------------------------------------------------------------------------
__global__ __launch_bounds__(128) void k_attn(const _Float16* __restrict__ Qh,
                                              const _Float16* __restrict__ Kh,
                                              const _Float16* __restrict__ Vt,
                                              _Float16* __restrict__ O,
                                              int S, int Spad) {
  __shared__ __align__(16) float    sS[4][16 * SMAXP];
  __shared__ __align__(16) _Float16 sP[4][16 * SMAXP];
  const int h = blockIdx.x, b = blockIdx.y;
  const int wave = threadIdx.x >> 5, lane = threadIdx.x & 31;
  const _Float16* Qb = Qh + (size_t)(b * kNH  + h)        * S * kDH;
  const _Float16* Kb = Kh + (size_t)(b * kNKV + (h >> 1)) * S * kDH;
  const _Float16* Vb = Vt + (size_t)(b * kNKV + (h >> 1)) * kDH * SMAXP;
  const int nT = S >> 4;                   // 9 (row) or 1 (col)

  for (int mt = wave; mt < nT; mt += 4) {
    const int m0 = mt * 16;
    v16h qf[4];
#pragma unroll
    for (int kk = 0; kk < 4; ++kk)
      qf[kk] = frag_ld(Qb + (size_t)m0 * kDH + kk * 32, kDH);

    for (int nt = 0; nt < nT; ++nt) {      // scores tile row: Q * K^T
      v8f acc = {};
#pragma unroll
      for (int kk = 0; kk < 4; ++kk) {
        v16h kf = frag_ld(Kb + (size_t)(nt * 16) * kDH + kk * 32, kDH);
        acc = wmma32(qf[kk], kf, acc);
      }
      int col = nt * 16 + (lane & 15);
      int rb  = (lane >> 4) << 3;
#pragma unroll
      for (int e = 0; e < 8; ++e) {
        int i = m0 + rb + e;
        float v = acc[e] * kScale;
        if (i >= kT && col >= kT && col > i) v = kNeg;
        sS[wave][(rb + e) * SMAXP + col] = v;
      }
    }
    // in-wave softmax (lanes 0-15, one row each; wave-synchronous, no barrier)
    if (lane < 16) {
      float* row = &sS[wave][lane * SMAXP];
      float mx = kNeg;
      for (int j = 0; j < S; ++j) mx = fmaxf(mx, row[j]);
      float sum = 0.f;
      for (int j = 0; j < S; ++j) { float e = __expf(row[j] - mx); row[j] = e; sum += e; }
      float inv = 1.f / sum;
      _Float16* pr = &sP[wave][lane * SMAXP];
      for (int j = 0; j < S;    ++j) pr[j] = (_Float16)(row[j] * inv);
      for (int j = S; j < Spad; ++j) pr[j] = (_Float16)0.f;
    }
    // O tile = P[16 x Spad] * V[Spad x 128]
    for (int dt = 0; dt < kDH / 16; ++dt) {
      v8f acc = {};
      for (int kk = 0; kk < Spad; kk += 32) {
        v16h pf = frag_ld(&sP[wave][kk], SMAXP);
        v16h vf = frag_ld(Vb + (size_t)(dt * 16) * SMAXP + kk, SMAXP);
        acc = wmma32(pf, vf, acc);
      }
      int col = dt * 16 + (lane & 15);
      int rb  = (lane >> 4) << 3;
#pragma unroll
      for (int e = 0; e < 8; ++e) {
        int s = m0 + rb + e;
        O[((size_t)(b * S + s) * kNH + h) * kDH + col] = (_Float16)acc[e];
      }
    }
  }
}

// ---------------------------------------------------------------------------
// Elementwise / norm / layout kernels
// ---------------------------------------------------------------------------
__global__ void k_cast(const float* __restrict__ in, _Float16* __restrict__ out) {
  size_t i = (size_t)blockIdx.x * 256 + threadIdx.x;
  out[i] = (_Float16)in[i];
}

__global__ void k_add(float* __restrict__ x, const float* __restrict__ y) {
  size_t i = (size_t)blockIdx.x * 256 + threadIdx.x;
  x[i] += y[i];
}

__global__ void k_copy(float* __restrict__ dst, const float* __restrict__ src) {
  size_t i = (size_t)blockIdx.x * 256 + threadIdx.x;
  dst[i] = src[i];
}

__global__ void k_silu_mul(_Float16* __restrict__ g, const _Float16* __restrict__ u) {
  size_t i = (size_t)blockIdx.x * 256 + threadIdx.x;
  float gv = (float)g[i], uv = (float)u[i];
  g[i] = (_Float16)(gv / (1.f + __expf(-gv)) * uv);
}

__global__ void k_concat(const float* __restrict__ lt, const float* __restrict__ dt,
                         float* __restrict__ x) {
  size_t i = (size_t)blockIdx.x * 256 + threadIdx.x;   // B*Srow*H
  int hh = (int)(i % kH);
  size_t bs = i / kH;
  int s = (int)(bs % kSrow), b = (int)(bs / kSrow);
  x[i] = (s < kT) ? lt[((size_t)b * kT + s) * kH + hh]
                  : dt[((size_t)b * kL + (s - kT)) * kH + hh];
}

__global__ void k_slice_rows(const float* __restrict__ x, float* __restrict__ out,
                             int S, int s0, int cnt) {
  size_t i = (size_t)blockIdx.x * 256 + threadIdx.x;   // B*cnt*H
  int hh = (int)(i % kH);
  size_t bj = i / kH;
  int j = (int)(bj % cnt), b = (int)(bj / cnt);
  out[i] = x[((size_t)b * S + s0 + j) * kH + hh];
}

// row-wise RMSNorm over H, f32 in -> f16 out; one block (8 waves) per row
__global__ __launch_bounds__(256) void k_rmsnorm(const float* __restrict__ x,
                                                 const float* __restrict__ w,
                                                 _Float16* __restrict__ out, int Hd) {
  __shared__ float red[8];
  const float* xr = x + (size_t)blockIdx.x * Hd;
  float ss = 0.f;
  for (int i = threadIdx.x; i < Hd; i += 256) { float v = xr[i]; ss += v * v; }
  for (int o = 16; o; o >>= 1) ss += __shfl_xor(ss, o, 32);
  if ((threadIdx.x & 31) == 0) red[threadIdx.x >> 5] = ss;
  __syncthreads();
  float tot = 0.f;
#pragma unroll
  for (int j = 0; j < 8; ++j) tot += red[j];
  float inv = rsqrtf(tot / Hd + kEps);
  _Float16* orow = out + (size_t)blockIdx.x * Hd;
  for (int i = threadIdx.x; i < Hd; i += 256) orow[i] = (_Float16)(xr[i] * inv * w[i]);
}

// per-head RMSNorm(DH=128) + RoPE; relayout [b][s][hn][d] -> [b][hn][s][d]
// 128 thr = 4 waves, one head-row per wave; lane owns d = lane,+32,+64,+96
// so the rotate-half partner (d, d+64) stays in-lane.
__global__ __launch_bounds__(128) void k_headnorm_rope(const _Float16* __restrict__ in,
                                                       const float* __restrict__ w,
                                                       _Float16* __restrict__ out,
                                                       int S, int HN, int use_rope) {
  int wave = threadIdx.x >> 5, lane = threadIdx.x & 31;
  int row = blockIdx.x * 4 + wave;                 // (b*S + s)*HN + hn
  int hn = row % HN;
  int bs = row / HN;
  int s = bs % S, b = bs / S;
  const _Float16* src = in + (size_t)row * kDH;
  float v0 = (float)src[lane],      v1 = (float)src[lane + 32];
  float v2 = (float)src[lane + 64], v3 = (float)src[lane + 96];
  float ss = v0 * v0 + v1 * v1 + v2 * v2 + v3 * v3;
  for (int o = 16; o; o >>= 1) ss += __shfl_xor(ss, o, 32);
  float inv = rsqrtf(ss * (1.f / kDH) + kEps);
  v0 *= inv * w[lane];      v1 *= inv * w[lane + 32];
  v2 *= inv * w[lane + 64]; v3 *= inv * w[lane + 96];
  if (use_rope) {
    float f0 = __expf(-kLnTheta * (float)lane        * (1.f / 64.f));
    float f1 = __expf(-kLnTheta * (float)(lane + 32) * (1.f / 64.f));
    float c0 = __cosf((float)s * f0), s0 = __sinf((float)s * f0);
    float c1 = __cosf((float)s * f1), s1 = __sinf((float)s * f1);
    float o0 = v0 * c0 - v2 * s0, o1 = v1 * c1 - v3 * s1;
    float o2 = v2 * c0 + v0 * s0, o3 = v3 * c1 + v1 * s1;
    v0 = o0; v1 = o1; v2 = o2; v3 = o3;
  }
  _Float16* dst = out + ((size_t)(b * HN + hn) * S + s) * kDH;
  dst[lane] = (_Float16)v0;      dst[lane + 32] = (_Float16)v1;
  dst[lane + 64] = (_Float16)v2; dst[lane + 96] = (_Float16)v3;
}

// V relayout: [b][s][kv][d] -> [b][kv][d][SMAXP] (N-major for PV B-fragments)
__global__ void k_vtrans(const _Float16* __restrict__ v, _Float16* __restrict__ vt, int S) {
  size_t idx = (size_t)blockIdx.x * 256 + threadIdx.x;  // B*S*NKV*DH
  int d = (int)(idx & 127);
  size_t r = idx >> 7;
  int kv = (int)(r % kNKV);
  size_t bs = r / kNKV;
  int s = (int)(bs % S), b = (int)(bs / S);
  vt[((size_t)(b * kNKV + kv) * kDH + d) * SMAXP + s] = v[idx];
}

// ---------------------------------------------------------------------------
// Host-side orchestration
// ---------------------------------------------------------------------------
struct LayerW {
  const float *in_ln, *q_norm, *k_norm, *post_ln;
  _Float16 *wq, *wk, *wv, *wo, *wg, *wu, *wd;
};

static void run_layer(float* x, int S, int Spad, int use_rope, const LayerW& w,
                      _Float16* h16, _Float16* q16, _Float16* k16, _Float16* v16b,
                      _Float16* Qh, _Float16* Kh, _Float16* Vt, _Float16* O16,
                      float* tmp32, _Float16* g16, _Float16* u16, hipStream_t stream) {
  const int tokens = kB * S;
  k_rmsnorm<<<dim3(tokens), 256, 0, stream>>>(x, w.in_ln, h16, kH);
  k_gemm<1><<<dim3((kNH * kDH) / GTN, tokens / GTM), 256, 0, stream>>>(h16, w.wq, nullptr, q16, tokens, kNH * kDH, kH);
  k_gemm<1><<<dim3((kNKV * kDH) / GTN, tokens / GTM), 256, 0, stream>>>(h16, w.wk, nullptr, k16, tokens, kNKV * kDH, kH);
  k_gemm<1><<<dim3((kNKV * kDH) / GTN, tokens / GTM), 256, 0, stream>>>(h16, w.wv, nullptr, v16b, tokens, kNKV * kDH, kH);
  k_headnorm_rope<<<dim3((tokens * kNH) / 4), 128, 0, stream>>>(q16, w.q_norm, Qh, S, kNH, use_rope);
  k_headnorm_rope<<<dim3((tokens * kNKV) / 4), 128, 0, stream>>>(k16, w.k_norm, Kh, S, kNKV, use_rope);
  k_vtrans<<<dim3(tokens * 4), 256, 0, stream>>>(v16b, Vt, S);
  k_attn<<<dim3(kNH, kB), 128, 0, stream>>>(Qh, Kh, Vt, O16, S, Spad);
  k_gemm<0><<<dim3(kH / GTN, tokens / GTM), 256, 0, stream>>>(O16, w.wo, tmp32, nullptr, tokens, kH, kNH * kDH);
  k_add<<<dim3((tokens * kH) / 256), 256, 0, stream>>>(x, tmp32);
  k_rmsnorm<<<dim3(tokens), 256, 0, stream>>>(x, w.post_ln, h16, kH);
  k_gemm<1><<<dim3(kFF / GTN, tokens / GTM), 256, 0, stream>>>(h16, w.wg, nullptr, g16, tokens, kFF, kH);
  k_gemm<1><<<dim3(kFF / GTN, tokens / GTM), 256, 0, stream>>>(h16, w.wu, nullptr, u16, tokens, kFF, kH);
  k_silu_mul<<<dim3((tokens * kFF) / 256), 256, 0, stream>>>(g16, u16);
  k_gemm<0><<<dim3(kH / GTN, tokens / GTM), 256, 0, stream>>>(g16, w.wd, tmp32, nullptr, tokens, kH, kFF);
  k_add<<<dim3((tokens * kH) / 256), 256, 0, stream>>>(x, tmp32);
}

extern "C" void kernel_launch(void* const* d_in, const int* in_sizes, int n_in,
                              void* d_out, int out_size, void* d_ws, size_t ws_size,
                              hipStream_t stream) {
  (void)in_sizes; (void)n_in; (void)out_size; (void)ws_size;
  const float* learned = (const float*)d_in[0];
  const float* dictt   = (const float*)d_in[1];
  // layer param dict order: in_ln,q,k,v,o,q_norm,k_norm,post_ln,gate,up,down
  const float* rp[11]; const float* cp[11];
  for (int i = 0; i < 11; ++i) { rp[i] = (const float*)d_in[2 + i]; cp[i] = (const float*)d_in[13 + i]; }

  char* p = (char*)d_ws;
  auto alloc = [&](size_t bytes) -> void* {
    void* r = (void*)p; p += (bytes + 255) & ~(size_t)255; return r;
  };
  const int tokR = kB * kSrow, tokC = kB * kT;
  float*    x     = (float*)   alloc((size_t)tokR * kH * 4);
  _Float16* h16   = (_Float16*)alloc((size_t)tokR * kH * 2);
  _Float16* q16   = (_Float16*)alloc((size_t)tokR * kNH * kDH * 2);
  _Float16* k16   = (_Float16*)alloc((size_t)tokR * kNKV * kDH * 2);
  _Float16* v16b  = (_Float16*)alloc((size_t)tokR * kNKV * kDH * 2);
  _Float16* Qh    = (_Float16*)alloc((size_t)tokR * kNH * kDH * 2);
  _Float16* Kh    = (_Float16*)alloc((size_t)tokR * kNKV * kDH * 2);
  _Float16* Vt    = (_Float16*)alloc((size_t)kB * kNKV * kDH * SMAXP * 2);
  _Float16* O16   = (_Float16*)alloc((size_t)tokR * kNH * kDH * 2);
  float*    tmp32 = (float*)   alloc((size_t)tokR * kH * 4);
  _Float16* g16   = (_Float16*)alloc((size_t)tokR * kFF * 2);
  _Float16* u16   = (_Float16*)alloc((size_t)tokR * kFF * 2);
  float*    x2    = (float*)   alloc((size_t)tokC * kH * 4);

  auto castw = [&](const float* src, size_t n) -> _Float16* {
    _Float16* dst = (_Float16*)alloc(n * 2);
    k_cast<<<dim3((unsigned)(n / 256)), 256, 0, stream>>>(src, dst);
    return dst;
  };
  auto make = [&](const float* const* pr) -> LayerW {
    LayerW w;
    w.in_ln = pr[0]; w.q_norm = pr[5]; w.k_norm = pr[6]; w.post_ln = pr[7];
    w.wq = castw(pr[1],  (size_t)kH * kNH * kDH);
    w.wk = castw(pr[2],  (size_t)kH * kNKV * kDH);
    w.wv = castw(pr[3],  (size_t)kH * kNKV * kDH);
    w.wo = castw(pr[4],  (size_t)kNH * kDH * kH);
    w.wg = castw(pr[8],  (size_t)kH * kFF);
    w.wu = castw(pr[9],  (size_t)kH * kFF);
    w.wd = castw(pr[10], (size_t)kFF * kH);
    return w;
  };
  LayerW rw = make(rp);
  LayerW cw = make(cp);

  // x = concat(learned, dict) along sequence
  k_concat<<<dim3((tokR * kH) / 256), 256, 0, stream>>>(learned, dictt, x);

  // row layer: S=144 (padded to 160), RoPE on
  run_layer(x, kSrow, SMAXP, 1, rw, h16, q16, k16, v16b, Qh, Kh, Vt, O16, tmp32, g16, u16, stream);

  float* outp = (float*)d_out;
  // upd_dict = row_out[:, T:, :]  -> second output slot
  k_slice_rows<<<dim3((kB * kL * kH) / 256), 256, 0, stream>>>(x, outp + (size_t)kB * kT * kH, kSrow, kT, kL);
  // col input = row_out[:, :T, :]
  k_slice_rows<<<dim3((tokC * kH) / 256), 256, 0, stream>>>(x, x2, kSrow, 0, kT);

  // col layer: S=16 (padded to 32), RoPE identity
  run_layer(x2, kT, 32, 0, cw, h16, q16, k16, v16b, Qh, Kh, Vt, O16, tmp32, g16, u16, stream);

  // col_out -> first output slot
  k_copy<<<dim3((tokC * kH) / 256), 256, 0, stream>>>(outp, x2);
}
